// ESMFoldSeedModel_66520453481117
// MI455X (gfx1250) — matrix operand order repaired and checked by reading, not compiled, not run on hardware
//
#include <hip/hip_runtime.h>

// ---------------------------------------------------------------------------
// ESMFold-seed forward on MI455X (gfx1250), wave32 + WMMA f32_16x16x32_f16.
// All heavy GEMMs (seq QKV/O/FFN, batched attention, pair FFN, distogram)
// go through v_wmma_f32_16x16x32_f16 with on-the-fly f32->f16 conversion.
// Pair FFN is fully fused (LN + GEMM1 + exact GELU + GEMM2 + residual + mask)
// so the 105MB pair tensor stays L2-resident instead of round-tripping HBM.
// V is produced transposed ([dim][token]) so every GEMM operand streams as
// contiguous b128 loads; A tiles are double-buffered in LDS (1 barrier/kstep).
// ---------------------------------------------------------------------------

typedef __attribute__((ext_vector_type(16))) _Float16 v16h;
typedef __attribute__((ext_vector_type(8)))  _Float16 v8h;
typedef __attribute__((ext_vector_type(4)))  _Float16 v4h;
typedef __attribute__((ext_vector_type(8)))  float    v8f;

#define BN 2
#define LSEQ 320
#define DM 512
#define NHD 8
#define HD 64
#define PD 128
#define FFH 2048
#define PFH 256
#define DISTB 64
#define PLDDTB 50
#define ATT_SCALE 0.125f   // HEAD_DIM^-0.5, HEAD_DIM=64
#define LN_EPS 1e-5f
#define INV_SQRT_PAIR 0.08838834764831845f  // 128^-0.5
#define LL (LSEQ * LSEQ)

static __device__ __forceinline__ float gelu_exact(float x) {
  return 0.5f * x * (1.0f + erff(x * 0.7071067811865475f));
}

// A fragment: 16x32 f16, contiguous k order (same as frag_b_nk). lanes 0-15: halves0-15->K0..15
//                                      lanes16-31: halves0-15->K16..31
// base points at (row0, k0) of an LDS tile, ldk in halves; 16B-aligned rows ->
// two ds_load_b128 per fragment.
static __device__ __forceinline__ v16h frag_a_lds(const _Float16* __restrict__ base,
                                                  int ldk, int lane) {
  const int m  = lane & 15;
  const int kb = (lane < 16) ? 0 : 16;
  const _Float16* p = base + m * ldk + kb;
  v8h lo = *(const v8h*)(p);
  v8h hi = *(const v8h*)(p + 8);
  v16h a;
#pragma unroll
  for (int h = 0; h < 8; ++h) { a[h] = lo[h]; a[8 + h] = hi[h]; }
  return a;
}

// B fragment (32x16 f16): lane -> column n; halves 0..15 -> K = h + (lane<16 ? 0 : 16).
// Source is a global f32 matrix stored [N][K] row-major: element (n,k) = W[n*ldk + k].
static __device__ __forceinline__ v16h frag_b_nk(const float* __restrict__ W, int ldk,
                                                 int n0, int k0, int lane) {
  const int n  = n0 + (lane & 15);
  const int kb = k0 + ((lane < 16) ? 0 : 16);
  const float* p = W + (size_t)n * ldk + kb;  // 16 consecutive f32, 16B-aligned
  v16h b;
#pragma unroll
  for (int q4 = 0; q4 < 4; ++q4) {
    float4 f = ((const float4*)p)[q4];
    b[q4 * 4 + 0] = (_Float16)f.x;
    b[q4 * 4 + 1] = (_Float16)f.y;
    b[q4 * 4 + 2] = (_Float16)f.z;
    b[q4 * 4 + 3] = (_Float16)f.w;
  }
  return b;
}

// ---------------------------------------------------------------------------
// Generic batched WMMA GEMM: C[b,h] = A[b,h] (MxK, f32) * B^T + bias
//  - B always [N][K] row-major f32, converted inline (contiguous b128 loads)
//  - ctrans: store C transposed ([N][M]); each lane's 8 accs are contiguous
//    in M -> two float4 global stores per tile.
//  - 128 threads = 4 waves; 64x64 tile; double-buffered A (1 barrier/kstep).
// M,N multiples of 64 (encoded in grid), K multiple of 32.
// ---------------------------------------------------------------------------
__global__ __launch_bounds__(128)
void gemm_wmma_kernel(const float* __restrict__ A, int lda, long long sAb, long long sAh,
                      const float* __restrict__ Bm, int ldb, long long sBb, long long sBh,
                      const float* __restrict__ bias,
                      float* __restrict__ C, int ldc, long long sCb, long long sCh,
                      int ctrans, int K, int nh) {
  __shared__ alignas(16) _Float16 As[2][64][40];   // double-buffered 64x32 tile
  const int tid  = threadIdx.x;
  const int wave = tid >> 5;
  const int lane = tid & 31;
  const int nlane = lane & 15;
  const int m0 = blockIdx.x * 64;
  const int n0 = blockIdx.y * 64;
  const int zb = blockIdx.z / nh;
  const int zh = blockIdx.z % nh;
  const float* Ab = A + zb * sAb + zh * sAh;
  const float* Bb = Bm + zb * sBb + zh * sBh;
  float*       Cb = C + zb * sCb + zh * sCh;

  // Cooperative A loader: thread owns row (tid>>1), 16-col half (tid&1).
  const int lr = tid >> 1;
  const int ch = (tid & 1) * 16;
  const float* arow = Ab + (size_t)(m0 + lr) * lda + ch;   // hoisted, 16B aligned

  v8f acc[4];
#pragma unroll
  for (int t = 0; t < 4; ++t) {
    float bv = bias ? bias[n0 + t * 16 + nlane] : 0.0f;
#pragma unroll
    for (int r = 0; r < 8; ++r) acc[t][r] = bv;
  }

  int buf = 0;
  for (int k0 = 0; k0 < K; k0 += 32) {
    const float4* s4 = (const float4*)(arow + k0);
    const float4 f0 = s4[0], f1 = s4[1], f2 = s4[2], f3 = s4[3];
    v8h h0, h1;
    h0[0] = (_Float16)f0.x; h0[1] = (_Float16)f0.y;
    h0[2] = (_Float16)f0.z; h0[3] = (_Float16)f0.w;
    h0[4] = (_Float16)f1.x; h0[5] = (_Float16)f1.y;
    h0[6] = (_Float16)f1.z; h0[7] = (_Float16)f1.w;
    h1[0] = (_Float16)f2.x; h1[1] = (_Float16)f2.y;
    h1[2] = (_Float16)f2.z; h1[3] = (_Float16)f2.w;
    h1[4] = (_Float16)f3.x; h1[5] = (_Float16)f3.y;
    h1[6] = (_Float16)f3.z; h1[7] = (_Float16)f3.w;
    _Float16* asrow = &As[buf][lr][ch];
    *(v8h*)(asrow)     = h0;
    *(v8h*)(asrow + 8) = h1;
    __syncthreads();   // single barrier per kstep (next store goes to other buf)
    const v16h av = frag_a_lds(&As[buf][wave * 16][0], 40, lane);
#pragma unroll
    for (int t = 0; t < 4; ++t) {
      const v16h bv = frag_b_nk(Bb, ldb, n0 + t * 16, k0, lane);
      acc[t] = __builtin_amdgcn_wmma_f32_16x16x32_f16(false, av, false, bv,
                                                      (short)0, acc[t], false, false);
    }
    buf ^= 1;
  }

  const int mrow = m0 + wave * 16 + ((lane < 16) ? 0 : 8);
  if (ctrans) {
#pragma unroll
    for (int t = 0; t < 4; ++t) {
      const int n = n0 + t * 16 + nlane;
      float* cp = Cb + (size_t)n * ldc + mrow;   // mrow multiple of 8 -> 16B aligned
      float4 lo, hi;
      lo.x = acc[t][0]; lo.y = acc[t][1]; lo.z = acc[t][2]; lo.w = acc[t][3];
      hi.x = acc[t][4]; hi.y = acc[t][5]; hi.z = acc[t][6]; hi.w = acc[t][7];
      *(float4*)(cp)     = lo;
      *(float4*)(cp + 4) = hi;
    }
  } else {
#pragma unroll
    for (int t = 0; t < 4; ++t) {
      const int n = n0 + t * 16 + nlane;
#pragma unroll
      for (int r = 0; r < 8; ++r) Cb[(size_t)(mrow + r) * ldc + n] = acc[t][r];
    }
  }
}

// ---------------------------------------------------------------------------
// Fused pair FFN: z = (z + W2*gelu(W1*LN(z)+b1)+b2) * pair_mask
// One block = 64 pair rows; LN in LDS(f16); two WMMA GEMMs (K=128, K=256).
// ---------------------------------------------------------------------------
__global__ __launch_bounds__(128)
void pair_ffn_kernel(float* __restrict__ z,
                     const float* __restrict__ g, const float* __restrict__ be,
                     const float* __restrict__ W1, const float* __restrict__ b1,
                     const float* __restrict__ W2, const float* __restrict__ b2,
                     const int* __restrict__ rm) {
  __shared__ alignas(16) _Float16 T[64][136];    // LN(z) tile, 64 x 128
  __shared__ alignas(16) _Float16 Hs[64][264];   // gelu hidden, 64 x 256
  __shared__ float mu[64], rs[64];
  const int tid  = threadIdx.x;
  const int wave = tid >> 5;
  const int lane = tid & 31;
  const int nlane = lane & 15;
  const int row0 = blockIdx.x * 64;

  if (tid < 64) {
    const float4* zr = (const float4*)(z + (size_t)(row0 + tid) * PD);
    float s = 0.f, s2 = 0.f;
#pragma unroll
    for (int d4 = 0; d4 < PD / 4; ++d4) {
      const float4 v = zr[d4];
      s  += v.x + v.y + v.z + v.w;
      s2 += v.x * v.x + v.y * v.y + v.z * v.z + v.w * v.w;
    }
    const float m = s * (1.0f / PD);
    const float var = s2 * (1.0f / PD) - m * m;
    mu[tid] = m;
    rs[tid] = rsqrtf(var + LN_EPS);
  }
  __syncthreads();
#pragma unroll
  for (int it = 0; it < 16; ++it) {             // 64 rows x 32 float4 / 128 threads
    const int e = tid + it * 128;
    const int r = e >> 5, c4 = e & 31;
    const float4 v  = *(const float4*)(z + (size_t)(row0 + r) * PD + c4 * 4);
    const float4 gg = *(const float4*)(g + c4 * 4);
    const float4 bb = *(const float4*)(be + c4 * 4);
    const float mm = mu[r], rr = rs[r];
    v4h o;
    o[0] = (_Float16)((v.x - mm) * rr * gg.x + bb.x);
    o[1] = (_Float16)((v.y - mm) * rr * gg.y + bb.y);
    o[2] = (_Float16)((v.z - mm) * rr * gg.z + bb.z);
    o[3] = (_Float16)((v.w - mm) * rr * gg.w + bb.w);
    *(v4h*)(&T[r][c4 * 4]) = o;
  }
  __syncthreads();

  const int mloc = wave * 16 + ((lane < 16) ? 0 : 8);

  // GEMM1: 16 rows x 256 cols, K=128, + GELU -> Hs
  for (int nt = 0; nt < 16; ++nt) {
    v8f acc;
    const float bb = b1[nt * 16 + nlane];
#pragma unroll
    for (int r = 0; r < 8; ++r) acc[r] = bb;
#pragma unroll
    for (int ks = 0; ks < 4; ++ks) {
      const v16h av = frag_a_lds(&T[wave * 16][ks * 32], 136, lane);
      const v16h bv = frag_b_nk(W1, PD, nt * 16, ks * 32, lane);
      acc = __builtin_amdgcn_wmma_f32_16x16x32_f16(false, av, false, bv,
                                                   (short)0, acc, false, false);
    }
#pragma unroll
    for (int r = 0; r < 8; ++r)
      Hs[mloc + r][nt * 16 + nlane] = (_Float16)gelu_exact(acc[r]);
  }
  __syncthreads();

  // GEMM2: 16 rows x 128 cols, K=256, + residual + pair mask (32-bit index math)
  for (int nt = 0; nt < 8; ++nt) {
    v8f acc;
    const float bb = b2[nt * 16 + nlane];
#pragma unroll
    for (int r = 0; r < 8; ++r) acc[r] = bb;
#pragma unroll
    for (int ks = 0; ks < 8; ++ks) {
      const v16h av = frag_a_lds(&Hs[wave * 16][ks * 32], 264, lane);
      const v16h bv = frag_b_nk(W2, PFH, nt * 16, ks * 32, lane);
      acc = __builtin_amdgcn_wmma_f32_16x16x32_f16(false, av, false, bv,
                                                   (short)0, acc, false, false);
    }
#pragma unroll
    for (int r = 0; r < 8; ++r) {
      const int row = row0 + mloc + r;
      const int b   = row / LL;
      const int rem = row - b * LL;
      const int i = rem / LSEQ;
      const int j = rem - i * LSEQ;
      const int n = nt * 16 + nlane;
      const size_t zi = (size_t)row * PD + n;
      const bool pm = (rm[b * LSEQ + i] != 0) && (rm[b * LSEQ + j] != 0);
      z[zi] = pm ? (z[zi] + acc[r]) : 0.0f;
    }
  }
}

// ---------------------------------------------------------------------------
// Fused (scale + optional pair bias + masks + softmax) per (b,h,q) row.
// logits in/out in place. zbias==nullptr -> plain encoder MHA.
// ---------------------------------------------------------------------------
__global__ __launch_bounds__(256)
void attn_softmax_kernel(float* __restrict__ logits, const int* __restrict__ rm,
                         const float* __restrict__ zbias, const float* __restrict__ bw,
                         int useQmask, float maskval) {
  __shared__ float red[256];
  const int bid = blockIdx.x;
  const int q = bid % LSEQ;
  const int h = (bid / LSEQ) % NHD;
  const int b = bid / (LSEQ * NHD);
  float* lrow = logits + (((size_t)(b * NHD + h) * LSEQ) + q) * LSEQ;
  const float* wh = zbias ? (bw + h * PD) : nullptr;
  const int tid = threadIdx.x;
  const bool qok = useQmask ? (rm[b * LSEQ + q] != 0) : true;

  float lv[2];
  int cnt = 0;
  for (int k = tid; k < LSEQ; k += 256) {
    float v = lrow[k] * ATT_SCALE;
    if (wh) {
      const float4* zr4 = (const float4*)(zbias + ((size_t)(b * LSEQ + q) * LSEQ + k) * PD);
      const float4* wh4 = (const float4*)wh;
      float acc = 0.f;
#pragma unroll 8
      for (int p4 = 0; p4 < PD / 4; ++p4) {
        const float4 a = zr4[p4], w = wh4[p4];
        acc += a.x * w.x + a.y * w.y + a.z * w.z + a.w * w.w;
      }
      v += acc;
    }
    if (!qok || (rm[b * LSEQ + k] == 0)) v = maskval;
    lv[cnt++] = v;
  }

  float mx = -3.4e38f;
  for (int c = 0; c < cnt; ++c) mx = fmaxf(mx, lv[c]);
  red[tid] = mx;
  __syncthreads();
  for (int st = 128; st >= 1; st >>= 1) {
    if (tid < st) red[tid] = fmaxf(red[tid], red[tid + st]);
    __syncthreads();
  }
  mx = red[0];
  __syncthreads();

  float sum = 0.f;
  for (int c = 0; c < cnt; ++c) { lv[c] = __expf(lv[c] - mx); sum += lv[c]; }
  red[tid] = sum;
  __syncthreads();
  for (int st = 128; st >= 1; st >>= 1) {
    if (tid < st) red[tid] += red[tid + st];
    __syncthreads();
  }
  const float inv = 1.0f / red[0];

  cnt = 0;
  for (int k = tid; k < LSEQ; k += 256) lrow[k] = lv[cnt++] * inv;
}

// ---------------------------------------------------------------------------
// Small elementwise / row kernels (float4 data paths, 32-bit indices)
// ---------------------------------------------------------------------------
__global__ void embed_ln_kernel(const int* __restrict__ aat, const float* __restrict__ aaE,
                                const float* __restrict__ posE,
                                const float* __restrict__ g, const float* __restrict__ be,
                                float* __restrict__ s) {
  const int row = blockIdx.x * blockDim.x + threadIdx.x;
  if (row >= BN * LSEQ) return;
  const int l = row % LSEQ;
  int a = aat[row];
  a = a < 0 ? 0 : (a > 20 ? 20 : a);
  const float4* ea = (const float4*)(aaE + (size_t)a * DM);
  const float4* ep = (const float4*)(posE + (size_t)l * DM);
  float sum = 0.f, s2 = 0.f;
  for (int d4 = 0; d4 < DM / 4; ++d4) {
    const float4 e4 = ea[d4], p4 = ep[d4];
    const float vx = e4.x + p4.x, vy = e4.y + p4.y, vz = e4.z + p4.z, vw = e4.w + p4.w;
    sum += vx + vy + vz + vw;
    s2 += vx * vx + vy * vy + vz * vz + vw * vw;
  }
  const float m = sum * (1.0f / DM);
  const float r = rsqrtf(s2 * (1.0f / DM) - m * m + LN_EPS);
  float4* y = (float4*)(s + (size_t)row * DM);
  const float4* g4 = (const float4*)g;
  const float4* b4 = (const float4*)be;
  for (int d4 = 0; d4 < DM / 4; ++d4) {
    const float4 e4 = ea[d4], p4 = ep[d4], gg = g4[d4], bb = b4[d4];
    float4 o;
    o.x = (e4.x + p4.x - m) * r * gg.x + bb.x;
    o.y = (e4.y + p4.y - m) * r * gg.y + bb.y;
    o.z = (e4.z + p4.z - m) * r * gg.z + bb.z;
    o.w = (e4.w + p4.w - m) * r * gg.w + bb.w;
    y[d4] = o;
  }
}

__global__ void ln_kernel(const float* __restrict__ in, float* __restrict__ out,
                          const float* __restrict__ g, const float* __restrict__ be,
                          int rows, int dim) {
  const int row = blockIdx.x * blockDim.x + threadIdx.x;
  if (row >= rows) return;
  const float4* x4 = (const float4*)(in + (size_t)row * dim);
  float4* y4 = (float4*)(out + (size_t)row * dim);
  const float4* g4 = (const float4*)g;
  const float4* b4 = (const float4*)be;
  const int n4 = dim >> 2;
  float s = 0.f, s2 = 0.f;
  for (int d = 0; d < n4; ++d) {
    const float4 v = x4[d];
    s  += v.x + v.y + v.z + v.w;
    s2 += v.x * v.x + v.y * v.y + v.z * v.z + v.w * v.w;
  }
  const float m = s / dim;
  const float r = rsqrtf(s2 / dim - m * m + LN_EPS);
  for (int d = 0; d < n4; ++d) {
    const float4 v = x4[d], gg = g4[d], bb = b4[d];
    float4 o;
    o.x = (v.x - m) * r * gg.x + bb.x;
    o.y = (v.y - m) * r * gg.y + bb.y;
    o.z = (v.z - m) * r * gg.z + bb.z;
    o.w = (v.w - m) * r * gg.w + bb.w;
    y4[d] = o;
  }
}

__global__ void add_inplace_kernel(float* __restrict__ dst, const float* __restrict__ src, int n4) {
  const int i = blockIdx.x * blockDim.x + threadIdx.x;
  if (i >= n4) return;
  const float4 a = ((const float4*)dst)[i];
  const float4 b = ((const float4*)src)[i];
  float4 o; o.x = a.x + b.x; o.y = a.y + b.y; o.z = a.z + b.z; o.w = a.w + b.w;
  ((float4*)dst)[i] = o;
}

__global__ void gelu_inplace_kernel(float* __restrict__ x, int n4) {
  const int i = blockIdx.x * blockDim.x + threadIdx.x;
  if (i >= n4) return;
  float4 v = ((const float4*)x)[i];
  v.x = gelu_exact(v.x); v.y = gelu_exact(v.y);
  v.z = gelu_exact(v.z); v.w = gelu_exact(v.w);
  ((float4*)x)[i] = v;
}

__global__ void mask_rows_kernel(float* __restrict__ s, const int* __restrict__ rm) {
  const int i = blockIdx.x * blockDim.x + threadIdx.x;   // over float4s of (640,512)
  if (i >= BN * LSEQ * (DM / 4)) return;
  if (rm[i / (DM / 4)] == 0) {
    float4 z4; z4.x = z4.y = z4.z = z4.w = 0.0f;
    ((float4*)s)[i] = z4;
  }
}

// z[b,i,j,p] = relposW[p][diff] + relposB[p] + left[b,i,p] + right[b,j,p]
__global__ void pair_init_kernel(float* __restrict__ z, const float* __restrict__ rw,
                                 const float* __restrict__ rb, const float* __restrict__ left,
                                 const float* __restrict__ right, const int* __restrict__ rm) {
  const int idx = blockIdx.x * blockDim.x + threadIdx.x;  // over float4s (6.55M)
  if (idx >= BN * LL * (PD / 4)) return;
  const int row = idx >> 5;
  const int p0 = (idx & 31) * 4;
  const int b = row / LL;
  const int rem = row - b * LL;
  const int i = rem / LSEQ;
  const int j = rem - i * LSEQ;
  const bool pm = (rm[b * LSEQ + i] != 0) && (rm[b * LSEQ + j] != 0);
  int diff = j - i;
  diff = diff < -32 ? -32 : (diff > 32 ? 32 : diff);
  diff += 33;
  if (!pm) diff = 0;
  const float4 rb4 = *(const float4*)(rb + p0);
  const float4 l4  = *(const float4*)(left + (size_t)(b * LSEQ + i) * PD + p0);
  const float4 r4  = *(const float4*)(right + (size_t)(b * LSEQ + j) * PD + p0);
  float4 o;
  o.x = rw[(p0 + 0) * 66 + diff] + rb4.x + l4.x + r4.x;
  o.y = rw[(p0 + 1) * 66 + diff] + rb4.y + l4.y + r4.y;
  o.z = rw[(p0 + 2) * 66 + diff] + rb4.z + l4.z + r4.z;
  o.w = rw[(p0 + 3) * 66 + diff] + rb4.w + l4.w + r4.w;
  ((float4*)z)[idx] = o;
}

// z += outer(left,right) * PAIR_DIM^-0.5
__global__ void outer_update_kernel(float* __restrict__ z, const float* __restrict__ left,
                                    const float* __restrict__ right) {
  const int idx = blockIdx.x * blockDim.x + threadIdx.x;  // over float4s
  if (idx >= BN * LL * (PD / 4)) return;
  const int row = idx >> 5;
  const int p0 = (idx & 31) * 4;
  const int b = row / LL;
  const int rem = row - b * LL;
  const int i = rem / LSEQ;
  const int j = rem - i * LSEQ;
  const float4 l4 = *(const float4*)(left + (size_t)(b * LSEQ + i) * PD + p0);
  const float4 r4 = *(const float4*)(right + (size_t)(b * LSEQ + j) * PD + p0);
  float4 v = ((const float4*)z)[idx];
  v.x += l4.x * r4.x * INV_SQRT_PAIR;
  v.y += l4.y * r4.y * INV_SQRT_PAIR;
  v.z += l4.z * r4.z * INV_SQRT_PAIR;
  v.w += l4.w * r4.w * INV_SQRT_PAIR;
  ((float4*)z)[idx] = v;
}

// Small-N heads (coord N=3, plddt N=50): one thread per (row,n) dot product.
__global__ void head_kernel(const float* __restrict__ in, const float* __restrict__ W,
                            const float* __restrict__ bias, float* __restrict__ out,
                            const int* __restrict__ rmOrNull, int rows, int N, int K) {
  const int idx = blockIdx.x * blockDim.x + threadIdx.x;
  if (idx >= rows * N) return;
  const int row = idx / N, n = idx % N;
  const float4* x4 = (const float4*)(in + (size_t)row * K);
  const float4* w4 = (const float4*)(W + (size_t)n * K);
  float acc = bias[n];
  for (int k4 = 0; k4 < (K >> 2); ++k4) {
    const float4 a = x4[k4], w = w4[k4];
    acc += a.x * w.x + a.y * w.y + a.z * w.z + a.w * w.w;
  }
  if (rmOrNull && rmOrNull[row] == 0) acc = 0.0f;
  out[idx] = acc;
}

// In-place symmetrize dist logits: out[b,i,j,:] = out[b,j,i,:] = 0.5*(a+b), i<j.
__global__ void symmetrize_kernel(float* __restrict__ d) {
  const int idx = blockIdx.x * blockDim.x + threadIdx.x;  // over float4s (3.28M)
  if (idx >= BN * LL * (DISTB / 4)) return;
  const int c0 = (idx & 15) * 4;
  const int row = idx >> 4;
  const int b = row / LL;
  const int rem = row - b * LL;
  const int i = rem / LSEQ;
  const int j = rem - i * LSEQ;
  if (i < j) {
    float4* p0 = (float4*)(d + (((size_t)b * LSEQ + i) * LSEQ + j) * DISTB + c0);
    float4* p1 = (float4*)(d + (((size_t)b * LSEQ + j) * LSEQ + i) * DISTB + c0);
    const float4 a = *p0, bb = *p1;
    float4 v;
    v.x = 0.5f * (a.x + bb.x); v.y = 0.5f * (a.y + bb.y);
    v.z = 0.5f * (a.z + bb.z); v.w = 0.5f * (a.w + bb.w);
    *p0 = v; *p1 = v;
  }
}

// ---------------------------------------------------------------------------
// Host orchestration
// ---------------------------------------------------------------------------
struct LinP { const float* w; const float* b; };
struct LnP  { const float* g; const float* b; };

extern "C" void kernel_launch(void* const* d_in, const int* in_sizes, int n_in,
                              void* d_out, int out_size, void* d_ws, size_t ws_size,
                              hipStream_t stream) {
  (void)in_sizes; (void)n_in; (void)out_size; (void)ws_size;

  // --- unpack inputs (JAX pytree leaf order: sorted dict keys) -------------
  int ix = 0;
  auto F = [&]() -> const float* { return (const float*)d_in[ix++]; };
  const int* aatype = (const int*)d_in[ix++];              // "aatype"
  const float* aa_embed = F();                             // params.aa_embed
  struct BlockP {
    LinP ff1, ff2, k, o, outer_left, outer_right;
    const float* pair_bias_w;
    LnP pair_ln;
    LinP pff1, pff2, q;
    LnP seq_ff_ln, seq_ln;
    LinP v;
  } blk[4];
  for (int t = 0; t < 4; ++t) {
    BlockP& B_ = blk[t];
    B_.ff1.b = F(); B_.ff1.w = F();
    B_.ff2.b = F(); B_.ff2.w = F();
    B_.k.b = F();   B_.k.w = F();
    B_.o.b = F();   B_.o.w = F();
    B_.outer_left.b = F();  B_.outer_left.w = F();
    B_.outer_right.b = F(); B_.outer_right.w = F();
    B_.pair_bias_w = F();
    B_.pair_ln.b = F(); B_.pair_ln.g = F();
    B_.pff1.b = F(); B_.pff1.w = F();
    B_.pff2.b = F(); B_.pff2.w = F();
    B_.q.b = F();    B_.q.w = F();
    B_.seq_ff_ln.b = F(); B_.seq_ff_ln.g = F();
    B_.seq_ln.b = F();    B_.seq_ln.g = F();
    B_.v.b = F();    B_.v.w = F();
  }
  LinP coord;  coord.b = F();  coord.w = F();
  LinP disto;  disto.b = F();  disto.w = F();
  struct EncP { LinP ff1, ff2, k; LnP ln1, ln2; LinP o, q, v; } enc[2];
  for (int t = 0; t < 2; ++t) {
    EncP& E = enc[t];
    E.ff1.b = F(); E.ff1.w = F();
    E.ff2.b = F(); E.ff2.w = F();
    E.k.b = F();   E.k.w = F();
    E.ln1.b = F(); E.ln1.g = F();
    E.ln2.b = F(); E.ln2.g = F();
    E.o.b = F();   E.o.w = F();
    E.q.b = F();   E.q.w = F();
    E.v.b = F();   E.v.w = F();
  }
  LnP out_ln;  out_ln.b = F();  out_ln.g = F();
  LinP p_left;  p_left.b = F();  p_left.w = F();
  LnP p_ln;     p_ln.b = F();    p_ln.g = F();
  LinP p_right; p_right.b = F(); p_right.w = F();
  LinP plddt;   plddt.b = F();   plddt.w = F();
  const float* pos_embed = F();
  LinP relpos;  relpos.b = F();  relpos.w = F();
  LnP state_in; state_in.b = F(); state_in.g = F();
  const int* rmask = (const int*)d_in[ix++];               // "residue_mask"

  // --- workspace bump allocator --------------------------------------------
  char* wsb = (char*)d_ws;
  size_t off = 0;
  auto alloc = [&](size_t nf) -> float* {
    float* p = (float*)(wsb + off);
    off = (off + nf * sizeof(float) + 255) & ~(size_t)255;
    return p;
  };
  const size_t NTOK = (size_t)BN * LSEQ;                   // 640
  const size_t NPAIR = (size_t)BN * LSEQ * LSEQ;           // 204800
  float* Z    = alloc(NPAIR * PD);                         // 104.9 MB
  float* S    = alloc(NTOK * DM);
  float* XN   = alloc(NTOK * DM);
  float* QB   = alloc(NTOK * DM);
  float* KB   = alloc(NTOK * DM);
  float* VT   = alloc(NTOK * DM);     // V stored transposed: [512 dims][640 tokens]
  float* OB   = alloc(NTOK * DM);
  float* HB   = alloc(NTOK * FFH);
  float* LOG  = alloc((size_t)BN * NHD * LSEQ * LSEQ);
  float* LEFT = alloc(NTOK * PD);
  float* RIGHT= alloc(NTOK * PD);
  float* SOUT = alloc(NTOK * DM);

  float* out = (float*)d_out;
  float* out_ca    = out;                                  // (2,320,3)
  float* out_dist  = out + NTOK * 3;                       // (2,320,320,64)
  float* out_plddt = out_dist + NPAIR * DISTB;             // (2,320,50)

  // --- launch helpers -------------------------------------------------------
  auto ew = [&](size_t n) { return dim3((unsigned)((n + 255) / 256)); };
  auto gemm = [&](const float* A, int lda, long long sAb, long long sAh,
                  const float* Bm, int ldb, long long sBb, long long sBh,
                  const float* bias, float* C, int ldc, long long sCb, long long sCh,
                  int ctrans, int M, int N, int K, int batches, int nh) {
    dim3 g((unsigned)(M / 64), (unsigned)(N / 64), (unsigned)batches);
    gemm_wmma_kernel<<<g, 128, 0, stream>>>(A, lda, sAb, sAh, Bm, ldb, sBb, sBh,
                                            bias, C, ldc, sCb, sCh, ctrans, K, nh);
  };
  auto gemm1 = [&](const float* A, const LinP& L_, float* C, int M, int N, int K) {
    gemm(A, K, 0, 0, L_.w, K, 0, 0, L_.b, C, N, 0, 0, 0, M, N, K, 1, 1);
  };
  auto lnr = [&](const float* in, float* o, const LnP& p, int rows, int dim) {
    ln_kernel<<<ew(rows), 256, 0, stream>>>(in, o, p.g, p.b, rows, dim);
  };
  auto addip = [&](float* dst, const float* src, size_t n) {
    add_inplace_kernel<<<ew(n / 4), 256, 0, stream>>>(dst, src, (int)(n / 4));
  };
  const long long sAB = (long long)LSEQ * DM;   // per-batch token stride
  const long long sLL = (long long)LSEQ * LSEQ;

  // mha: XN already holds LN(s); zbias!=null -> pair attention
  auto mha = [&](const LinP& q, const LinP& k, const LinP& v, const LinP& o,
                 const float* zbias, const float* bw, float maskval, int useQ) {
    gemm1(XN, q, QB, (int)NTOK, DM, DM);
    gemm1(XN, k, KB, (int)NTOK, DM, DM);
    // V projection stored transposed: VT[n=h*64+d][m=b*320+l], ld = 640
    gemm(XN, DM, 0, 0, v.w, DM, 0, 0, v.b, VT, (int)NTOK, 0, 0, 1,
         (int)NTOK, DM, DM, 1, 1);
    // logits[b,h] = Q[b,h] (320x64) * K[b,h]^T  -> (320x320)
    gemm(QB, DM, sAB, HD, KB, DM, sAB, HD, nullptr,
         LOG, LSEQ, (long long)NHD * sLL, sLL, 0, LSEQ, LSEQ, HD, BN * NHD, NHD);
    attn_softmax_kernel<<<BN * NHD * LSEQ, 256, 0, stream>>>(LOG, rmask, zbias, bw,
                                                             useQ, maskval);
    // O[b,h] = attn (320x320) * V[b,h]; V as [N=dim][K=token] slice of VT:
    // element (n,k) = VT[(h*64+n)*640 + b*320 + k]  -> ldb=640, sBb=320, sBh=64*640
    gemm(LOG, LSEQ, (long long)NHD * sLL, sLL, VT, (int)NTOK, LSEQ,
         (long long)HD * NTOK, nullptr,
         OB, DM, sAB, HD, 0, LSEQ, HD, LSEQ, BN * NHD, NHD);
    gemm1(OB, o, XN, (int)NTOK, DM, DM);
    addip(S, XN, NTOK * DM);
  };
  auto ffn = [&](const LnP& lnp, const LinP& f1, const LinP& f2) {
    lnr(S, XN, lnp, (int)NTOK, DM);
    gemm1(XN, f1, HB, (int)NTOK, FFH, DM);
    gelu_inplace_kernel<<<ew(NTOK * FFH / 4), 256, 0, stream>>>(HB, (int)(NTOK * FFH / 4));
    gemm1(HB, f2, XN, (int)NTOK, DM, FFH);
    addip(S, XN, NTOK * DM);
  };

  // --- forward pass ---------------------------------------------------------
  embed_ln_kernel<<<ew(NTOK), 256, 0, stream>>>(aatype, aa_embed, pos_embed,
                                                state_in.g, state_in.b, S);
  for (int e = 0; e < 2; ++e) {
    lnr(S, XN, enc[e].ln1, (int)NTOK, DM);
    mha(enc[e].q, enc[e].k, enc[e].v, enc[e].o, nullptr, nullptr, -1e9f, 0);
    ffn(enc[e].ln2, enc[e].ff1, enc[e].ff2);
  }
  mask_rows_kernel<<<ew(NTOK * DM / 4), 256, 0, stream>>>(S, rmask);

  // pair init: z = relpos[diff] + b + left_i + right_j ; z = LN(z)
  gemm1(S, p_left,  LEFT,  (int)NTOK, PD, DM);
  gemm1(S, p_right, RIGHT, (int)NTOK, PD, DM);
  pair_init_kernel<<<ew(NPAIR * PD / 4), 256, 0, stream>>>(Z, relpos.w, relpos.b,
                                                           LEFT, RIGHT, rmask);
  lnr(Z, Z, p_ln, (int)NPAIR, PD);

  for (int t = 0; t < 4; ++t) {
    const BlockP& B_ = blk[t];
    lnr(S, XN, B_.seq_ln, (int)NTOK, DM);
    mha(B_.q, B_.k, B_.v, B_.o, Z, B_.pair_bias_w, -10000.0f, 1);
    ffn(B_.seq_ff_ln, B_.ff1, B_.ff2);
    gemm1(S, B_.outer_left,  LEFT,  (int)NTOK, PD, DM);
    gemm1(S, B_.outer_right, RIGHT, (int)NTOK, PD, DM);
    outer_update_kernel<<<ew(NPAIR * PD / 4), 256, 0, stream>>>(Z, LEFT, RIGHT);
    pair_ffn_kernel<<<(unsigned)(NPAIR / 64), 128, 0, stream>>>(
        Z, B_.pair_ln.g, B_.pair_ln.b, B_.pff1.w, B_.pff1.b, B_.pff2.w, B_.pff2.b, rmask);
    mask_rows_kernel<<<ew(NTOK * DM / 4), 256, 0, stream>>>(S, rmask);
  }

  // heads
  lnr(S, SOUT, out_ln, (int)NTOK, DM);
  head_kernel<<<ew(NTOK * 3), 256, 0, stream>>>(SOUT, coord.w, coord.b, out_ca,
                                                rmask, (int)NTOK, 3, DM);
  gemm1(Z, disto, out_dist, (int)NPAIR, DISTB, PD);
  symmetrize_kernel<<<ew(NPAIR * DISTB / 4), 256, 0, stream>>>(out_dist);
  head_kernel<<<ew(NTOK * PLDDTB), 256, 0, stream>>>(SOUT, plddt.w, plddt.b, out_plddt,
                                                     nullptr, (int)NTOK, PLDDTB, DM);
}
